// WeatherGNN_53815940218922
// MI455X (gfx1250) — matrix operand order repaired and checked by reading, not compiled, not run on hardware
//
#include <hip/hip_runtime.h>
#include <hip/hip_bf16.h>
#include <stdint.h>

#define HID     256
#define IN_CH   16
#define OUT_CH  16
#define NGRAPH  64
#define NLAYERS 3
#define KCHEB   3
#define EPS     1e-5f

#if __has_builtin(__builtin_amdgcn_global_load_async_to_lds_b128)
#define HAVE_ASYNC_LDS 1
#else
#define HAVE_ASYNC_LDS 0
#endif

// ---------------------------------------------------------------------------
// WMMA types (CDNA5 gfx1250, wave32): 16x16x32 bf16 -> f32
// ---------------------------------------------------------------------------
typedef __attribute__((ext_vector_type(16))) __bf16 v16bf;
typedef __attribute__((ext_vector_type(8)))  float  v8f;

// exact pointee type of the async-LDS builtin (from hipcc diagnostic):
// 'int __attribute__((vector_size(16)))'
typedef int vint4 __attribute__((vector_size(16)));
typedef __attribute__((address_space(1))) vint4 gvint4;  // global
typedef __attribute__((address_space(3))) vint4 lvint4;  // LDS

union Frag32 {
    v16bf v;
    uint4 q[2];
};

static __device__ __forceinline__ unsigned pack2_bf16(float a, float b) {
    __bf16 ha = (__bf16)a, hb = (__bf16)b;
    unsigned short ua = __builtin_bit_cast(unsigned short, ha);
    unsigned short ub = __builtin_bit_cast(unsigned short, hb);
    return (unsigned)ua | ((unsigned)ub << 16);
}

static __device__ __forceinline__ float sigmoidf(float x) {
    return 1.0f / (1.0f + __expf(-x));
}

static __device__ __forceinline__ void wait_async_zero() {
#if HAVE_ASYNC_LDS
#if __has_builtin(__builtin_amdgcn_s_wait_asynccnt)
    __builtin_amdgcn_s_wait_asynccnt(0);
#else
    asm volatile("s_wait_asynccnt 0x0" ::: "memory");
#endif
#endif
}

// ---------------------------------------------------------------------------
// Weight pre-swizzle: fp32 [256][256] row-major (k,n)  ->  bf16 packed frags
// Layout: [mat][kb(8)][ct(16)][lane(32)][j(8)] dwords, each dword = 2 bf16.
// B-matrix (32x16, K x N) lane layout (ISA 7.12.2, dense analog of sparse B):
//   lane: n = ct*16 + (lane&15); khalf = lane>>4
//   VGPR j holds K = kb*32 + (j>>2)*16 + khalf*8 + 2*(j&3), +1
// One K-step tile (kb fixed) is 16*32*8 dwords = 16KB, linear in memory.
// ---------------------------------------------------------------------------
__global__ void pack_weights_kernel(const float* __restrict__ W,
                                    unsigned* __restrict__ out, int nmat) {
    size_t idx = (size_t)blockIdx.x * blockDim.x + threadIdx.x;
    size_t total = (size_t)nmat * 8 * 16 * 32 * 8;
    if (idx >= total) return;
    int j    = (int)(idx & 7);
    int lane = (int)((idx >> 3) & 31);
    int ct   = (int)((idx >> 8) & 15);
    int kb   = (int)((idx >> 12) & 7);
    int m    = (int)(idx >> 15);
    int n = ct * 16 + (lane & 15);
    int k = kb * 32 + (j >> 2) * 16 + (lane >> 4) * 8 + 2 * (j & 3);
    const float* Wm = W + (size_t)m * HID * HID;
    out[idx] = pack2_bf16(Wm[(size_t)k * HID + n], Wm[(size_t)(k + 1) * HID + n]);
}

// ---------------------------------------------------------------------------
// WMMA GEMM: C[N,256] = concat_k(A_k[N,256]) @ packedB  (K = nsrc*256)
// Block: 256 threads = 8 waves; tile 64 rows x 256 cols.
// Wave w: rowtile rt = w>>1 (16 rows), col-half ch = w&1 (8 col tiles).
//
// Pipeline (one s_barrier per K-step, double-buffered LDS):
//   - B tile for step kt+1 (16KB) is staged global->LDS with
//     global_load_async_to_lds_b128 (ASYNCcnt) while step kt's WMMAs run;
//     all 8 waves then share one LDS copy (4x less L2 traffic) and B
//     fragments become low-latency ds_load_b128 pairs.
//   - A tile for step kt+1 is prefetched to registers, converted to bf16,
//     committed to LDS at the top of the next step.
// ---------------------------------------------------------------------------
__global__ __launch_bounds__(256) void gemm_wmma_kernel(
        const float* __restrict__ A0, const float* __restrict__ A1,
        const float* __restrict__ A2, const unsigned* __restrict__ Bp,
        float* __restrict__ Cout, int N, int nsrc) {
    __shared__ __align__(16) unsigned lA[2][64 * 16];    // 2 x 4KB  (A tile)
    __shared__ __align__(16) unsigned lB[2][16 * 32 * 8];// 2 x 16KB (B tile)

    const int tid  = threadIdx.x;
    const int wave = tid >> 5;
    const int lane = tid & 31;
    const int row0 = blockIdx.x * 64;
    const int rt = wave >> 1;
    const int ch = wave & 1;

    // staging thread mapping for A: thread -> (row r, 8-float group q)
    const int sr = tid >> 2;
    const int sq = tid & 3;
    const bool srow_ok = (row0 + sr) < N;
    const float* srcPtr[3] = {A0, A1, A2};

    // B tile staging: 16KB per K-step; each thread moves 4 x 16B chunks
    auto stageB = [&](int kt, unsigned* ldst) {
        const unsigned* gsrc = Bp + (size_t)kt * (16 * 32 * 8);
#pragma unroll
        for (int it = 0; it < 4; it++) {
            int off = it * 1024 + tid * 4;  // dword offset
#if HAVE_ASYNC_LDS
            __builtin_amdgcn_global_load_async_to_lds_b128(
                (gvint4*)(gsrc + off), (lvint4*)(ldst + off), 0, 0);
#else
            *(uint4*)(ldst + off) = *(const uint4*)(gsrc + off);
#endif
        }
    };

    v8f acc[8];
#pragma unroll
    for (int c = 0; c < 8; c++)
#pragma unroll
        for (int r = 0; r < 8; r++) acc[c][r] = 0.0f;

    const int nkt = nsrc * 8;

    // ---- preloop: stage step-0 B into lB[0]; prefetch step-0 A into regs ----
    stageB(0, lB[0]);
    float4 f0 = {0.f, 0.f, 0.f, 0.f}, f1 = {0.f, 0.f, 0.f, 0.f};
    if (srow_ok) {
        const float4* gp =
            (const float4*)(srcPtr[0] + (size_t)(row0 + sr) * HID + sq * 8);
        f0 = gp[0];
        f1 = gp[1];
    }

    unsigned buf = 0;
    for (int kt = 0; kt < nkt; kt++) {
        // commit staged A registers for step kt into LDS buffer `buf`
        {
            unsigned* dp = &lA[buf][sr * 16 + sq * 4];
            dp[0] = pack2_bf16(f0.x, f0.y);
            dp[1] = pack2_bf16(f0.z, f0.w);
            dp[2] = pack2_bf16(f1.x, f1.y);
            dp[3] = pack2_bf16(f1.z, f1.w);
        }
        wait_async_zero();   // B writes into lB[buf] (issued last step) done
        __syncthreads();     // all waves' A stores + B stages visible

        // kick off step kt+1 staging (overlaps WMMA chain below)
        float4 n0 = {0.f, 0.f, 0.f, 0.f}, n1 = {0.f, 0.f, 0.f, 0.f};
        if (kt + 1 < nkt) {
            stageB(kt + 1, lB[buf ^ 1u]);
            if (srow_ok) {
                int kc = (kt + 1) >> 3, kb = (kt + 1) & 7;
                const float4* gp = (const float4*)(srcPtr[kc] +
                                                   (size_t)(row0 + sr) * HID +
                                                   kb * 32 + sq * 8);
                n0 = gp[0];
                n1 = gp[1];
            }
        }

        // A fragment from LDS (16-bit 16x32 layout, ISA 7.12.2)
        Frag32 a;
        {
            int m = lane & 15;
            int khalf = lane >> 4;
            const uint4* lp =
                (const uint4*)&lA[buf][(rt * 16 + m) * 16 + khalf * 4];
            a.q[0] = lp[0];  // K = khalf*8 .. +7
            a.q[1] = lp[2];  // K = 16 + khalf*8 .. +7 (+32B)
        }

        // 8 B fragments from LDS + 8 WMMAs
#pragma unroll
        for (int c = 0; c < 8; c++) {
            int ct = ch * 8 + c;
            Frag32 b;
            const uint4* bp =
                (const uint4*)&lB[buf][((size_t)ct * 32 + lane) * 8];
            b.q[0] = bp[0];
            b.q[1] = bp[1];
            acc[c] = __builtin_amdgcn_wmma_f32_16x16x32_bf16(
                false, a.v, false, b.v, (short)0, acc[c], false, false);
        }

        f0 = n0;
        f1 = n1;
        buf ^= 1u;
    }

    // C/D layout: VGPR r -> (M = r + (lane>=16 ? 8 : 0), N = lane&15)
    {
        int n = lane & 15;
        int rbase = (lane >> 4) * 8;
#pragma unroll
        for (int c = 0; c < 8; c++) {
            int col = (ch * 8 + c) * 16 + n;
#pragma unroll
            for (int r = 0; r < 8; r++) {
                int grow = row0 + rt * 16 + rbase + r;
                if (grow < N) Cout[(size_t)grow * HID + col] = acc[c][r];
            }
        }
    }
}

// ---------------------------------------------------------------------------
// Graph preprocessing
// ---------------------------------------------------------------------------
__global__ void degree_kernel(const int* __restrict__ src, float* __restrict__ deg, int E) {
    int i = blockIdx.x * blockDim.x + threadIdx.x;
    if (i < E) atomicAdd(&deg[src[i]], 1.0f);
}

__global__ void norm_kernel(const int* __restrict__ src, const int* __restrict__ dst,
                            const float* __restrict__ deg, float* __restrict__ nrm, int E) {
    int i = blockIdx.x * blockDim.x + threadIdx.x;
    if (i >= E) return;
    float ds = deg[src[i]], dd = deg[dst[i]];
    float a = (ds > 0.f) ? rsqrtf(fmaxf(ds, 1.f)) : 0.f;
    float b = (dd > 0.f) ? rsqrtf(fmaxf(dd, 1.f)) : 0.f;
    nrm[i] = -(a * b);
}

__global__ void counts_kernel(const int* __restrict__ batch, float* __restrict__ cnt, int N) {
    int i = blockIdx.x * blockDim.x + threadIdx.x;
    if (i < N) atomicAdd(&cnt[batch[i]], 1.0f);
}

// ---------------------------------------------------------------------------
// Input projection: h = x_t @ W[0:16] + ct  (ct = tf @ W[16:272] + b, shared)
// ---------------------------------------------------------------------------
__global__ void timeproj_kernel(const float* __restrict__ time_w,
                                const float* __restrict__ time_b,
                                const float* __restrict__ inproj_w,
                                const float* __restrict__ inproj_b,
                                float tfrac, float* __restrict__ ct) {
    int c = threadIdx.x;
    float acc = inproj_b[c];
    for (int k = 0; k < HID; k++) {
        float tf = tfrac * time_w[k] + time_b[k];
        acc += tf * inproj_w[(size_t)(IN_CH + k) * HID + c];
    }
    ct[c] = acc;
}

__global__ void inproj_kernel(const float* __restrict__ xt,
                              const float* __restrict__ W,
                              const float* __restrict__ ct,
                              float* __restrict__ h, int N) {
    int n = blockIdx.x, c = threadIdx.x;
    __shared__ float sx[IN_CH];
    if (c < IN_CH) sx[c] = xt[(size_t)n * IN_CH + c];
    __syncthreads();
    float acc = ct[c];
    for (int k = 0; k < IN_CH; k++) acc += sx[k] * W[(size_t)k * HID + c];
    h[(size_t)n * HID + c] = acc;
}

// ---------------------------------------------------------------------------
// Sparse propagation: out[dst] += norm_e * in[src]   (one block per edge)
// ---------------------------------------------------------------------------
__global__ void prop_kernel(const float* __restrict__ in, float* __restrict__ out,
                            const int* __restrict__ src, const int* __restrict__ dst,
                            const float* __restrict__ nrm, int E) {
    int e = blockIdx.x;
    int c = threadIdx.x;
    int s = src[e], d = dst[e];
    float w = nrm[e];
    float v = in[(size_t)s * HID + c] * w;
    atomicAdd(&out[(size_t)d * HID + c], v);
}

__global__ void cheb2_kernel(float* __restrict__ tx2, const float* __restrict__ tx0, size_t n) {
    size_t i = (size_t)blockIdx.x * blockDim.x + threadIdx.x;
    if (i < n) tx2[i] = 2.0f * tx2[i] - tx0[i];
}

// ---------------------------------------------------------------------------
// Epilogue: h = LayerNorm(relu(cheb_out + bias) + h) * g + b
// ---------------------------------------------------------------------------
__global__ void ln_kernel(const float* __restrict__ cheb_out,
                          const float* __restrict__ cb, float* __restrict__ h,
                          const float* __restrict__ g, const float* __restrict__ b,
                          int N) {
    int n = blockIdx.x, c = threadIdx.x;
    __shared__ float red[HID];
    float v = cheb_out[(size_t)n * HID + c] + cb[c];
    v = fmaxf(v, 0.0f) + h[(size_t)n * HID + c];
    red[c] = v;
    __syncthreads();
    for (int s = 128; s > 0; s >>= 1) {
        if (c < s) red[c] += red[c + s];
        __syncthreads();
    }
    float mu = red[0] * (1.0f / HID);
    __syncthreads();
    float d = v - mu;
    red[c] = d * d;
    __syncthreads();
    for (int s = 128; s > 0; s >>= 1) {
        if (c < s) red[c] += red[c + s];
        __syncthreads();
    }
    float var = red[0] * (1.0f / HID);
    h[(size_t)n * HID + c] = d * rsqrtf(var + EPS) * g[c] + b[c];
}

// ---------------------------------------------------------------------------
// Pooling: pooled[batch[n]] += node_out[n] + outproj_b; later /= counts
// ---------------------------------------------------------------------------
__global__ void pool_kernel(const float* __restrict__ node_out,
                            const float* __restrict__ outb,
                            const int* __restrict__ batch,
                            float* __restrict__ pooled, int N) {
    int n = blockIdx.x, c = threadIdx.x;
    int b = batch[n];
    atomicAdd(&pooled[(size_t)b * HID + c], node_out[(size_t)n * HID + c] + outb[c]);
}

__global__ void seqdiv_kernel(float* __restrict__ seq, const float* __restrict__ cnt) {
    int g = blockIdx.x;  // 0 .. T*64-1
    int b = g & (NGRAPH - 1);
    float inv = 1.0f / fmaxf(cnt[b], 1.0f);
    seq[(size_t)g * HID + threadIdx.x] *= inv;
}

// ---------------------------------------------------------------------------
// LSTM (64 graphs, HID=256)
// ---------------------------------------------------------------------------
__global__ void lstm_gates_kernel(const float* __restrict__ xt,
                                  const float* __restrict__ hprev,
                                  const float* __restrict__ wih,
                                  const float* __restrict__ whh,
                                  const float* __restrict__ bih,
                                  const float* __restrict__ bhh,
                                  float* __restrict__ gates) {
    int b = blockIdx.x, gq = blockIdx.y, c = threadIdx.x;
    int g = gq * 256 + c;
    __shared__ float sx[HID], sh[HID];
    sx[c] = xt[(size_t)b * HID + c];
    sh[c] = hprev[(size_t)b * HID + c];
    __syncthreads();
    float acc = bih[g] + bhh[g];
    const float* wi = wih + (size_t)g * HID;
    const float* wh = whh + (size_t)g * HID;
    for (int k = 0; k < HID; k++) acc += sx[k] * wi[k] + sh[k] * wh[k];
    gates[(size_t)b * 1024 + g] = acc;
}

__global__ void lstm_update_kernel(const float* __restrict__ gates,
                                   float* __restrict__ hstate,
                                   float* __restrict__ cstate) {
    int b = blockIdx.x, j = threadIdx.x;
    float gi = gates[(size_t)b * 1024 + j];
    float gf = gates[(size_t)b * 1024 + 256 + j];
    float gg = gates[(size_t)b * 1024 + 512 + j];
    float go = gates[(size_t)b * 1024 + 768 + j];
    float c = sigmoidf(gf) * cstate[(size_t)b * HID + j] + sigmoidf(gi) * tanhf(gg);
    cstate[(size_t)b * HID + j] = c;
    hstate[(size_t)b * HID + j] = sigmoidf(go) * tanhf(c);
}

// ---------------------------------------------------------------------------
// MLP head
// ---------------------------------------------------------------------------
__global__ void mlp1_kernel(const float* __restrict__ fin, const float* __restrict__ W,
                            const float* __restrict__ b1, float* __restrict__ z1) {
    int b = blockIdx.x, o = blockIdx.y * 256 + threadIdx.x;
    __shared__ float s[HID];
    s[threadIdx.x] = fin[(size_t)b * HID + threadIdx.x];
    __syncthreads();
    float acc = b1[o];
    for (int k = 0; k < HID; k++) acc += s[k] * W[(size_t)k * 512 + o];
    z1[(size_t)b * 512 + o] = fmaxf(acc, 0.0f);
}

__global__ void mlp2_kernel(const float* __restrict__ z1, const float* __restrict__ W,
                            const float* __restrict__ b2, float* __restrict__ z2) {
    int b = blockIdx.x, c = threadIdx.x;
    __shared__ float s[512];
    s[c] = z1[(size_t)b * 512 + c];
    s[c + 256] = z1[(size_t)b * 512 + 256 + c];
    __syncthreads();
    float acc = b2[c];
    for (int k = 0; k < 512; k++) acc += s[k] * W[(size_t)k * HID + c];
    z2[(size_t)b * HID + c] = fmaxf(acc, 0.0f);
}

__global__ void mlp3_kernel(const float* __restrict__ z2, const float* __restrict__ W,
                            const float* __restrict__ b3, float* __restrict__ out) {
    int idx = blockIdx.x * 256 + threadIdx.x;  // 64*16 = 1024
    int b = idx >> 4, o = idx & 15;
    float acc = b3[o];
    for (int k = 0; k < HID; k++) acc += z2[(size_t)b * HID + k] * W[(size_t)k * OUT_CH + o];
    out[idx] = acc;
}

// ---------------------------------------------------------------------------
// Host driver
// ---------------------------------------------------------------------------
extern "C" void kernel_launch(void* const* d_in, const int* in_sizes, int n_in,
                              void* d_out, int out_size, void* d_ws, size_t ws_size,
                              hipStream_t stream) {
    const float* x         = (const float*)d_in[0];
    const int*   edge      = (const int*)d_in[1];
    const int*   batch     = (const int*)d_in[2];
    const float* time_w    = (const float*)d_in[3];
    const float* time_b    = (const float*)d_in[4];
    const float* inproj_w  = (const float*)d_in[5];
    const float* inproj_b  = (const float*)d_in[6];
    const float* cheb_w    = (const float*)d_in[7];
    const float* cheb_b    = (const float*)d_in[8];
    const float* ln_g      = (const float*)d_in[9];
    const float* ln_b      = (const float*)d_in[10];
    const float* outproj_w = (const float*)d_in[11];
    const float* outproj_b = (const float*)d_in[12];
    const float* lstm_wih  = (const float*)d_in[13];
    const float* lstm_whh  = (const float*)d_in[14];
    const float* lstm_bih  = (const float*)d_in[15];
    const float* lstm_bhh  = (const float*)d_in[16];
    const float* mlp1_w    = (const float*)d_in[17];
    const float* mlp1_b    = (const float*)d_in[18];
    const float* mlp2_w    = (const float*)d_in[19];
    const float* mlp2_b    = (const float*)d_in[20];
    const float* mlp3_w    = (const float*)d_in[21];
    const float* mlp3_b    = (const float*)d_in[22];

    const int N = in_sizes[2];
    const int E = in_sizes[1] / 2;
    const int T = in_sizes[0] / (N * IN_CH);
    const int* srcI = edge;
    const int* dstI = edge + E;

    // --- workspace carve-out ---
    uint8_t* p = (uint8_t*)d_ws;
    auto carve = [&](size_t bytes) -> void* {
        void* r = (void*)p;
        p += (bytes + 255) & ~(size_t)255;
        return r;
    };
    const size_t NH = (size_t)N * HID;
    float*    h        = (float*)carve(NH * 4);
    float*    tx1      = (float*)carve(NH * 4);
    float*    tx2      = (float*)carve(NH * 4);
    float*    tmp      = (float*)carve(NH * 4);
    float*    deg      = (float*)carve((size_t)N * 4);
    float*    nrm      = (float*)carve((size_t)E * 4);
    float*    cnt      = (float*)carve(NGRAPH * 4);
    unsigned* packCheb = (unsigned*)carve((size_t)NLAYERS * KCHEB * 32768 * 4);
    unsigned* packOut  = (unsigned*)carve((size_t)32768 * 4);
    float*    seq      = (float*)carve((size_t)T * NGRAPH * HID * 4);
    float*    hstate   = (float*)carve((size_t)NGRAPH * HID * 4);
    float*    cstate   = (float*)carve((size_t)NGRAPH * HID * 4);
    float*    gates    = (float*)carve((size_t)NGRAPH * 1024 * 4);
    float*    z1       = (float*)carve((size_t)NGRAPH * 512 * 4);
    float*    z2       = (float*)carve((size_t)NGRAPH * HID * 4);
    float*    ctbuf    = (float*)carve(HID * 4);

    // --- preprocessing ---
    (void)hipMemsetAsync(deg, 0, (size_t)N * 4, stream);
    (void)hipMemsetAsync(cnt, 0, NGRAPH * 4, stream);
    (void)hipMemsetAsync(seq, 0, (size_t)T * NGRAPH * HID * 4, stream);
    (void)hipMemsetAsync(hstate, 0, (size_t)NGRAPH * HID * 4, stream);
    (void)hipMemsetAsync(cstate, 0, (size_t)NGRAPH * HID * 4, stream);

    degree_kernel<<<(E + 255) / 256, 256, 0, stream>>>(srcI, deg, E);
    norm_kernel<<<(E + 255) / 256, 256, 0, stream>>>(srcI, dstI, deg, nrm, E);
    counts_kernel<<<(N + 255) / 256, 256, 0, stream>>>(batch, cnt, N);

    {
        size_t tc = (size_t)NLAYERS * KCHEB * 32768;
        pack_weights_kernel<<<(unsigned)((tc + 255) / 256), 256, 0, stream>>>(
            cheb_w, packCheb, NLAYERS * KCHEB);
        pack_weights_kernel<<<(32768 + 255) / 256, 256, 0, stream>>>(
            outproj_w, packOut, 1);
    }

    const int gemmGrid = (N + 63) / 64;

    // --- temporal loop ---
    for (int t = 0; t < T; t++) {
        timeproj_kernel<<<1, 256, 0, stream>>>(time_w, time_b, inproj_w,
                                               inproj_b, (float)t / (float)T, ctbuf);
        inproj_kernel<<<N, 256, 0, stream>>>(x + (size_t)t * N * IN_CH,
                                             inproj_w, ctbuf, h, N);
        for (int L = 0; L < NLAYERS; L++) {
            (void)hipMemsetAsync(tx1, 0, NH * 4, stream);
            prop_kernel<<<E, 256, 0, stream>>>(h, tx1, srcI, dstI, nrm, E);
            (void)hipMemsetAsync(tx2, 0, NH * 4, stream);
            prop_kernel<<<E, 256, 0, stream>>>(tx1, tx2, srcI, dstI, nrm, E);
            cheb2_kernel<<<(unsigned)((NH + 255) / 256), 256, 0, stream>>>(tx2, h, NH);

            gemm_wmma_kernel<<<gemmGrid, 256, 0, stream>>>(
                h, tx1, tx2, packCheb + (size_t)L * KCHEB * 32768, tmp, N, 3);
            ln_kernel<<<N, 256, 0, stream>>>(tmp, cheb_b + L * HID, h,
                                             ln_g + L * HID, ln_b + L * HID, N);
        }
        gemm_wmma_kernel<<<gemmGrid, 256, 0, stream>>>(
            h, nullptr, nullptr, packOut, tmp, N, 1);
        pool_kernel<<<N, 256, 0, stream>>>(tmp, outproj_b, batch,
                                           seq + (size_t)t * NGRAPH * HID, N);
    }

    seqdiv_kernel<<<T * NGRAPH, 256, 0, stream>>>(seq, cnt);

    // --- LSTM ---
    for (int t = 0; t < T; t++) {
        lstm_gates_kernel<<<dim3(NGRAPH, 4), 256, 0, stream>>>(
            seq + (size_t)t * NGRAPH * HID, hstate, lstm_wih, lstm_whh,
            lstm_bih, lstm_bhh, gates);
        lstm_update_kernel<<<NGRAPH, 256, 0, stream>>>(gates, hstate, cstate);
    }

    // --- MLP head ---
    mlp1_kernel<<<dim3(NGRAPH, 2), 256, 0, stream>>>(hstate, mlp1_w, mlp1_b, z1);
    mlp2_kernel<<<NGRAPH, 256, 0, stream>>>(z1, mlp2_w, mlp2_b, z2);
    mlp3_kernel<<<4, 256, 0, stream>>>(z2, mlp3_w, mlp3_b, (float*)d_out);
}
